// GroupedQueryAttention_51505247814017
// MI455X (gfx1250) — matrix operand order repaired and checked by reading, not compiled, not run on hardware
//
#include <hip/hip_runtime.h>

// GQA forward for MI455X (gfx1250, wave32, WMMA + Tensor Data Mover).
// Pipeline: cvt->bf16 (weights pre-transposed), NT WMMA GEMM qkv proj, RoPE,
// flash-attention (WMMA, TDM double-buffered tiles), NT WMMA out proj.

#define BB   2
#define TT   2048
#define CC   1024
#define HH   16
#define KVHN 4
#define DD   64
#define HALF 32
#define BT   (BB * TT)   // 4096

typedef __attribute__((ext_vector_type(16))) __bf16 v16bf;
typedef __attribute__((ext_vector_type(8)))  float  v8f;
typedef __attribute__((ext_vector_type(4)))  unsigned int u32x4;
typedef __attribute__((ext_vector_type(8)))  int          i32x8;
typedef __attribute__((ext_vector_type(4)))  int          i32x4;

#if defined(__HIP_DEVICE_COMPILE__) && __has_builtin(__builtin_amdgcn_tensor_load_to_lds) && __has_builtin(__builtin_amdgcn_s_wait_tensorcnt)
#define HAVE_TDM 1
#else
#define HAVE_TDM 0
#endif

union FragB { v16bf v; unsigned int u[8]; };

__device__ __forceinline__ unsigned short f2bf(float f) {
  union { float f; unsigned int u; } v; v.f = f;
  unsigned int r = v.u + 0x7FFFu + ((v.u >> 16) & 1u);   // round-to-nearest-even
  return (unsigned short)(r >> 16);
}

#if HAVE_TDM
// Issue a 2D TDM tile load: tile_h rows of tile_w bf16 elements, row stride
// stride_elems, into contiguous LDS at lds_addr. D# per CDNA5 ISA ch.8.
__device__ __forceinline__ void tdm_load_2d(unsigned lds_addr, const void* gptr,
                                            unsigned tile_w, unsigned tile_h,
                                            unsigned stride_elems) {
  unsigned long long ga = (unsigned long long)(size_t)gptr;
  u32x4 g0;
  g0[0] = 1u;                                      // count=1, user mode
  g0[1] = lds_addr;                                // group0[63:32] lds_addr
  g0[2] = (unsigned)(ga & 0xFFFFFFFFu);            // global_addr[31:0]
  g0[3] = (unsigned)((ga >> 32) & 0x01FFFFFFu) | 0x80000000u; // addr[56:32] | type=2
  i32x8 g1;
  g1[0] = 0x00010000;                              // wg_mask=0, data_size=1 (2B)
  g1[1] = (int)((tile_w & 0xFFFFu) << 16);         // tensor_dim0 lo16
  g1[2] = (int)((tile_w >> 16) | ((tile_h & 0xFFFFu) << 16)); // dim0 hi | dim1 lo
  g1[3] = (int)((tile_h >> 16) | (tile_w << 16));  // dim1 hi | tile_dim0
  g1[4] = (int)(tile_h & 0xFFFFu);                 // tile_dim1 | tile_dim2=0
  g1[5] = (int)stride_elems;                       // tensor_dim0_stride lo32
  g1[6] = 0;                                       // stride0 hi | stride1 lo
  g1[7] = 0;
  i32x4 gz4 = {0, 0, 0, 0};
  i32x8 gz8 = {0, 0, 0, 0, 0, 0, 0, 0};
  __builtin_amdgcn_tensor_load_to_lds(g0, g1, gz4, gz4, gz8, 0);
}
#endif

// ---------------- elementwise converts ----------------
__global__ void cvt_bf16_kernel(const float* __restrict__ in,
                                unsigned short* __restrict__ out, int n) {
  int i = blockIdx.x * blockDim.x + threadIdx.x;
  if (i < n) out[i] = f2bf(in[i]);
}

// W[K][N] fp32 -> Wt[N][K] bf16 (NT layout for the GEMM)
__global__ void cvt_bf16_t_kernel(const float* __restrict__ in,
                                  unsigned short* __restrict__ out, int K, int N) {
  int idx = blockIdx.x * blockDim.x + threadIdx.x;
  if (idx >= K * N) return;
  int k = idx % K, n = idx / K;
  out[(size_t)n * K + k] = f2bf(in[(size_t)k * N + n]);
}

// vf [BT, KVH*64] fp32 -> vt[(b*KVH+kvh)*64 + d][T] bf16 (transposed per head)
__global__ void cvt_vt_kernel(const float* __restrict__ in,
                              unsigned short* __restrict__ out) {
  int idx = blockIdx.x * blockDim.x + threadIdx.x;
  int total = BB * KVHN * DD * TT;
  if (idx >= total) return;
  int t   = idx % TT;
  int row = idx / TT;                 // (bi*KVH + kvh)*64 + d
  int d   = row % DD;
  int kvh = (row / DD) % KVHN;
  int bi  = row / (DD * KVHN);
  out[idx] = f2bf(in[((size_t)(bi * TT + t)) * (KVHN * DD) + kvh * DD + d]);
}

// RoPE on [BT, heads*64] fp32 -> bf16 (same layout)
__global__ void rope_cvt_kernel(const float* __restrict__ in,
                                const float* __restrict__ cosT,
                                const float* __restrict__ sinT,
                                unsigned short* __restrict__ out, int heads) {
  int idx = blockIdx.x * blockDim.x + threadIdx.x;
  int total = BT * heads * HALF;
  if (idx >= total) return;
  int d  = idx % HALF;
  int h  = (idx / HALF) % heads;
  int bt = idx / (HALF * heads);
  int t  = bt % TT;
  int base = (bt * heads + h) * DD;
  float x1 = in[base + d], x2 = in[base + d + HALF];
  float c = cosT[t * HALF + d], s = sinT[t * HALF + d];
  out[base + d]        = f2bf(x1 * c - x2 * s);
  out[base + d + HALF] = f2bf(x2 * c + x1 * s);
}

// ---------------- NT bf16 WMMA GEMM: C[MxN] = A[MxK] @ Bt[NxK]^T ----------------
// 256 threads (8 waves), tile 128(M) x 128(N), K-step 32, double-buffered LDS.
__global__ __launch_bounds__(256) void gemm_bf16_nt_kernel(
    const unsigned short* __restrict__ A, const unsigned short* __restrict__ Bt,
    float* __restrict__ C, int M, int N, int K) {
  __shared__ unsigned short As[2][128][32];   // [m][k]
  __shared__ unsigned short Bs[2][128][32];   // [n][k]
  int tid = threadIdx.x;
  int lane = tid & 31, w = tid >> 5;
  int lhi = (lane >> 4) & 1, ln = lane & 15;
  int m0 = blockIdx.y * 128, n0 = blockIdx.x * 128;
  int wrow = (w & 3) * 32, wcol = (w >> 2) * 64;

  int nsteps = K / 32;

  auto stage = [&](int st, int buf) {
    int kk = st * 32;
#if HAVE_TDM
    if (w == 0) {
      tdm_load_2d((unsigned)(size_t)&As[buf][0][0], A  + (size_t)m0 * K + kk, 32, 128, (unsigned)K);
      tdm_load_2d((unsigned)(size_t)&Bs[buf][0][0], Bt + (size_t)n0 * K + kk, 32, 128, (unsigned)K);
    }
#else
    int r = tid >> 1, c0 = (tid & 1) * 16;
    const unsigned int* as = (const unsigned int*)(A + (size_t)(m0 + r) * K + kk + c0);
    unsigned int* ad = (unsigned int*)(&As[buf][r][c0]);
    const unsigned int* bs = (const unsigned int*)(Bt + (size_t)(n0 + r) * K + kk + c0);
    unsigned int* bd = (unsigned int*)(&Bs[buf][r][c0]);
#pragma unroll
    for (int j = 0; j < 8; ++j) { ad[j] = as[j]; bd[j] = bs[j]; }
#endif
  };
  auto stage_wait = [&]() {
#if HAVE_TDM
    if (w == 0) __builtin_amdgcn_s_wait_tensorcnt(0);
#endif
  };

  v8f acc[2][4] = {};

  stage(0, 0);
  stage_wait();
  __syncthreads();

  for (int st = 0; st < nsteps; ++st) {
    int buf = st & 1;
    if (st + 1 < nsteps) stage(st + 1, buf ^ 1);   // overlap next tile with compute

    FragB a[2];  // A 16x32 frag: m = ln, k = j + lhi*16
#pragma unroll
    for (int mf = 0; mf < 2; ++mf)
#pragma unroll
      for (int j2 = 0; j2 < 8; ++j2) {
        int j = 2 * j2;
        int k = j + (lhi ? 16 : 0);
        a[mf].u[j2] = *(const unsigned int*)(&As[buf][wrow + mf * 16 + ln][k]);
      }
#pragma unroll
    for (int nf = 0; nf < 4; ++nf) {
      FragB b;  // B 32x16 frag: n = ln, k = j + lhi*16
#pragma unroll
      for (int j2 = 0; j2 < 8; ++j2) {
        int k = 2 * j2 + (lhi ? 16 : 0);
        b.u[j2] = *(const unsigned int*)(&Bs[buf][wcol + nf * 16 + ln][k]);
      }
#pragma unroll
      for (int mf = 0; mf < 2; ++mf)
        acc[mf][nf] = __builtin_amdgcn_wmma_f32_16x16x32_bf16(
            false, a[mf].v, false, b.v, (short)0, acc[mf][nf], false, false);
    }
    stage_wait();
    __syncthreads();
  }

#pragma unroll
  for (int mf = 0; mf < 2; ++mf)
#pragma unroll
    for (int nf = 0; nf < 4; ++nf)
#pragma unroll
      for (int i = 0; i < 8; ++i) {
        int row = m0 + wrow + mf * 16 + i + lhi * 8;
        C[(size_t)row * N + n0 + wcol + nf * 16 + ln] = acc[mf][nf][i];
      }
}

// ---------------- flash attention (causal, GQA) ----------------
// grid = (T/64, B*H); block = 128 (4 waves x 16 q-rows). 32 keys per step,
// TDM double-buffered K/V tiles.
__global__ __launch_bounds__(128) void gqa_attn_kernel(
    const unsigned short* __restrict__ Q,   // [B,T,H,64] bf16 (post-RoPE)
    const unsigned short* __restrict__ Kg,  // [B,T,KVH,64] bf16 (post-RoPE)
    const unsigned short* __restrict__ Vt,  // [(B*KVH)*64, T] bf16 (transposed)
    unsigned short* __restrict__ Y) {       // [B,T,H,64] bf16
  __shared__ unsigned short Ks[2][32][64];     // [k][d]
  __shared__ unsigned short Vs[2][64][32];     // [d][k]
  __shared__ unsigned short Ps[4][16][32];     // per-wave P bounce

  int tid = threadIdx.x;
  int lane = tid & 31, w = tid >> 5;
  int lhi = (lane >> 4) & 1, ln = lane & 15;
  int q0 = blockIdx.x * 64;
  int bh = blockIdx.y;
  int bi = bh / HH, h = bh % HH;
  int kvh = h / 4;                             // N_GROUPS = 4
  const float scale = 0.125f;                  // 1/sqrt(64)

  const unsigned short* kbase_ptr = Kg + ((size_t)bi * TT * KVHN + kvh) * DD;
  const unsigned short* vbase_ptr = Vt + (size_t)(bi * KVHN + kvh) * DD * TT;

  auto stage = [&](int st, int buf) {
    int kb = st * 32;
#if HAVE_TDM
    if (w == 0) {
      tdm_load_2d((unsigned)(size_t)&Ks[buf][0][0], kbase_ptr + (size_t)kb * (KVHN * DD),
                  64, 32, KVHN * DD);
      tdm_load_2d((unsigned)(size_t)&Vs[buf][0][0], vbase_ptr + kb, 32, 64, TT);
    }
#else
    {
      int r = tid >> 2, c0 = (tid & 3) * 16;
      const unsigned int* src = (const unsigned int*)(kbase_ptr + (size_t)(kb + r) * (KVHN * DD) + c0);
      unsigned int* dst = (unsigned int*)(&Ks[buf][r][c0]);
#pragma unroll
      for (int j = 0; j < 8; ++j) dst[j] = src[j];
    }
    {
      int d = tid >> 1, c0 = (tid & 1) * 16;
      const unsigned int* src = (const unsigned int*)(vbase_ptr + (size_t)d * TT + kb + c0);
      unsigned int* dst = (unsigned int*)(&Vs[buf][d][c0]);
#pragma unroll
      for (int j = 0; j < 8; ++j) dst[j] = src[j];
    }
#endif
  };
  auto stage_wait = [&]() {
#if HAVE_TDM
    if (w == 0) __builtin_amdgcn_s_wait_tensorcnt(0);
#endif
  };

  // Q fragments (two K=32 halves of the head dim), direct from global
  int qrow = q0 + w * 16 + ln;
  const unsigned short* qp = Q + ((size_t)(bi * TT + qrow) * HH + h) * DD;
  FragB qf[2];
#pragma unroll
  for (int f = 0; f < 2; ++f)
#pragma unroll
    for (int j2 = 0; j2 < 8; ++j2) {
      int j = 2 * j2;
      int d = j + (lhi ? 16 : 0) + 32 * f;
      qf[f].u[j2] = *(const unsigned int*)(qp + d);
    }

  v8f o[4] = {};
  float mrow[8], lrow[8];
#pragma unroll
  for (int i = 0; i < 8; ++i) { mrow[i] = -3.0e38f; lrow[i] = 0.f; }

  int nsteps = q0 / 32 + 2;  // causal: keys up to q0+63
  stage(0, 0);
  stage_wait();
  __syncthreads();

  for (int st = 0; st < nsteps; ++st) {
    int buf = st & 1;
    int kb = st * 32;
    if (st + 1 < nsteps) stage(st + 1, buf ^ 1);   // overlap with compute

    // S[16q x 32k] = Q @ K^T
    v8f sacc[2] = {};
#pragma unroll
    for (int kt = 0; kt < 2; ++kt)
#pragma unroll
      for (int f = 0; f < 2; ++f) {
        FragB b;
#pragma unroll
        for (int j2 = 0; j2 < 8; ++j2) {
          int d = 2 * j2 + (lhi ? 16 : 0) + 32 * f;
          b.u[j2] = *(const unsigned int*)(&Ks[buf][kt * 16 + ln][d]);
        }
        sacc[kt] = __builtin_amdgcn_wmma_f32_16x16x32_bf16(
            false, qf[f].v, false, b.v, (short)0, sacc[kt], false, false);
      }

    // online softmax (rows split 8-per-lane across 16-lane halves)
    float pv[2][8], rmax[8];
#pragma unroll
    for (int i = 0; i < 8; ++i) {
      int row = q0 + w * 16 + i + lhi * 8;
      float best = -3.0e38f;
#pragma unroll
      for (int kt = 0; kt < 2; ++kt) {
        int col = kb + kt * 16 + ln;
        float s = sacc[kt][i] * scale;
        if (col > row) s = -3.0e38f;
        pv[kt][i] = s;
        best = fmaxf(best, s);
      }
      rmax[i] = best;
    }
#pragma unroll
    for (int mm = 1; mm < 16; mm <<= 1)
#pragma unroll
      for (int i = 0; i < 8; ++i)
        rmax[i] = fmaxf(rmax[i], __shfl_xor(rmax[i], mm, 32));

    float alpha[8], rsum[8];
#pragma unroll
    for (int i = 0; i < 8; ++i) {
      float mn = fmaxf(mrow[i], rmax[i]);
      alpha[i] = __expf(mrow[i] - mn);
      mrow[i] = mn;
      float sum = 0.f;
#pragma unroll
      for (int kt = 0; kt < 2; ++kt) {
        float p = __expf(pv[kt][i] - mn);
        pv[kt][i] = p;
        sum += p;
      }
      rsum[i] = sum;
    }
#pragma unroll
    for (int mm = 1; mm < 16; mm <<= 1)
#pragma unroll
      for (int i = 0; i < 8; ++i)
        rsum[i] += __shfl_xor(rsum[i], mm, 32);
#pragma unroll
    for (int i = 0; i < 8; ++i) lrow[i] = lrow[i] * alpha[i] + rsum[i];
#pragma unroll
    for (int nt = 0; nt < 4; ++nt)
#pragma unroll
      for (int i = 0; i < 8; ++i) o[nt][i] *= alpha[i];

    // bounce P (accum layout) -> LDS -> A-fragment layout
#pragma unroll
    for (int kt = 0; kt < 2; ++kt)
#pragma unroll
      for (int i = 0; i < 8; ++i)
        Ps[w][i + lhi * 8][kt * 16 + ln] = f2bf(pv[kt][i]);
    __syncthreads();

    FragB pf;
#pragma unroll
    for (int j2 = 0; j2 < 8; ++j2) {
      int j = 2 * j2;
      int k = j + (lhi ? 16 : 0);
      pf.u[j2] = *(const unsigned int*)(&Ps[w][ln][k]);
    }
#pragma unroll
    for (int nt = 0; nt < 4; ++nt) {
      FragB b;
#pragma unroll
      for (int j2 = 0; j2 < 8; ++j2) {
        int k = 2 * j2 + (lhi ? 16 : 0);
        b.u[j2] = *(const unsigned int*)(&Vs[buf][nt * 16 + ln][k]);
      }
      o[nt] = __builtin_amdgcn_wmma_f32_16x16x32_bf16(
          false, pf.v, false, b.v, (short)0, o[nt], false, false);
    }
    stage_wait();
    __syncthreads();
  }

  // normalize and store bf16 [B,T,H,64]
#pragma unroll
  for (int nt = 0; nt < 4; ++nt)
#pragma unroll
    for (int i = 0; i < 8; ++i) {
      int row = q0 + w * 16 + i + lhi * 8;
      float val = o[nt][i] / lrow[i];
      Y[((size_t)(bi * TT + row) * HH + h) * DD + nt * 16 + ln] = f2bf(val);
    }
}

// ---------------- launch ----------------
extern "C" void kernel_launch(void* const* d_in, const int* in_sizes, int n_in,
                              void* d_out, int out_size, void* d_ws, size_t ws_size,
                              hipStream_t stream) {
  const float* x    = (const float*)d_in[0];
  const float* cosT = (const float*)d_in[1];
  const float* sinT = (const float*)d_in[2];
  const float* Wq   = (const float*)d_in[3];
  const float* Wk   = (const float*)d_in[4];
  const float* Wv   = (const float*)d_in[5];
  const float* Wo   = (const float*)d_in[6];
  float* out = (float*)d_out;

  char* p = (char*)d_ws;
  auto alloc = [&](size_t bytes) {
    char* r = p;
    p += (bytes + 255) & ~(size_t)255;
    return r;
  };
  unsigned short* xb  = (unsigned short*)alloc((size_t)BT * CC * 2);
  unsigned short* Wqt = (unsigned short*)alloc((size_t)CC * 1024 * 2);  // [N][K]
  unsigned short* Wkt = (unsigned short*)alloc((size_t)CC * 256 * 2);
  unsigned short* Wvt = (unsigned short*)alloc((size_t)CC * 256 * 2);
  unsigned short* Wot = (unsigned short*)alloc((size_t)1024 * CC * 2);
  float* qf = (float*)alloc((size_t)BT * 1024 * 4);
  float* kf = (float*)alloc((size_t)BT * 256 * 4);
  float* vf = (float*)alloc((size_t)BT * 256 * 4);
  unsigned short* qb = (unsigned short*)alloc((size_t)BT * 1024 * 2);
  unsigned short* kb = (unsigned short*)alloc((size_t)BT * 256 * 2);
  unsigned short* vt = (unsigned short*)alloc((size_t)BT * 256 * 2);
  unsigned short* yb = (unsigned short*)alloc((size_t)BT * 1024 * 2);

  // converts (weights transposed to NT layout)
  cvt_bf16_kernel<<<(BT * CC + 255) / 256, 256, 0, stream>>>(x, xb, BT * CC);
  cvt_bf16_t_kernel<<<(CC * 1024 + 255) / 256, 256, 0, stream>>>(Wq, Wqt, CC, 1024);
  cvt_bf16_t_kernel<<<(CC * 256 + 255) / 256, 256, 0, stream>>>(Wk, Wkt, CC, 256);
  cvt_bf16_t_kernel<<<(CC * 256 + 255) / 256, 256, 0, stream>>>(Wv, Wvt, CC, 256);
  cvt_bf16_t_kernel<<<(1024 * CC + 255) / 256, 256, 0, stream>>>(Wo, Wot, 1024, CC);

  // QKV projections (fp32 accumulate out)
  gemm_bf16_nt_kernel<<<dim3(1024 / 128, BT / 128), 256, 0, stream>>>(xb, Wqt, qf, BT, 1024, CC);
  gemm_bf16_nt_kernel<<<dim3(256 / 128,  BT / 128), 256, 0, stream>>>(xb, Wkt, kf, BT, 256,  CC);
  gemm_bf16_nt_kernel<<<dim3(256 / 128,  BT / 128), 256, 0, stream>>>(xb, Wvt, vf, BT, 256,  CC);

  // RoPE (q,k) + transposed convert (v)
  {
    int n = BT * HH * HALF;
    rope_cvt_kernel<<<(n + 255) / 256, 256, 0, stream>>>(qf, cosT, sinT, qb, HH);
    n = BT * KVHN * HALF;
    rope_cvt_kernel<<<(n + 255) / 256, 256, 0, stream>>>(kf, cosT, sinT, kb, KVHN);
    n = BB * KVHN * DD * TT;
    cvt_vt_kernel<<<(n + 255) / 256, 256, 0, stream>>>(vf, vt);
  }

  // flash attention
  gqa_attn_kernel<<<dim3(TT / 64, BB * HH), 128, 0, stream>>>(qb, kb, vt, yb);

  // output projection straight to d_out
  gemm_bf16_nt_kernel<<<dim3(1024 / 128, BT / 128), 256, 0, stream>>>(yb, Wot, out, BT, CC, 1024);
}
